// RN9_54820962566534
// MI455X (gfx1250) — matrix operand order repaired and checked by reading, not compiled, not run on hardware
//
#include <hip/hip_runtime.h>

// Relation-network forward, algebraically restructured:
//   qe = relu(q @ W3 + b3)
//   u  = v * qe (broadcast over objects)
//   A  = u @ W1                         (linearity: pairs@W1 = A_i + A_j)
//   x[n,i] = sum_j relu(A[n,i] + A[n,j] + b1)
//   out = relu(x @ W2 + b2)
// All GEMM operands pre-converted to bf16 (weights transposed to [n][k]);
// GEMM loop: double-buffered async global->LDS staging (ASYNCcnt pipelined,
// s_wait_asynccnt 5 leaves next tile in flight) feeding
// v_wmma_f32_16x16x32_bf16 with fp32 accumulation.

#define NN 32
#define BB 36
#define DD 2048
#define QQ 1024

typedef __bf16 bf16_t;
typedef bf16_t bf16x2 __attribute__((ext_vector_type(2)));
typedef bf16_t v16bf  __attribute__((ext_vector_type(16)));
typedef float  v8f    __attribute__((ext_vector_type(8)));
typedef int    v4i    __attribute__((ext_vector_type(4)));

union Frag {
    v16bf v;
    float4 f[2];
};

// 16-byte global -> LDS copy; async (ASYNCcnt / direct-to-LDS) when available.
__device__ __forceinline__ void copy16(const void* g, void* l) {
#if __has_builtin(__builtin_amdgcn_global_load_async_to_lds_b128)
    __builtin_amdgcn_global_load_async_to_lds_b128((v4i*)g, (v4i*)l, 0, 0);
#else
    *reinterpret_cast<float4*>(l) = *reinterpret_cast<const float4*>(g);
#endif
}

template <int Ncnt>
__device__ __forceinline__ void copy_wait() {
#if __has_builtin(__builtin_amdgcn_global_load_async_to_lds_b128)
#if __has_builtin(__builtin_amdgcn_s_wait_asynccnt)
    __builtin_amdgcn_s_wait_asynccnt(Ncnt);
#else
    if (Ncnt == 0) { asm volatile("s_wait_asynccnt 0x0" ::: "memory"); }
    else           { asm volatile("s_wait_asynccnt 0x5" ::: "memory"); }
#endif
#endif
}

// C(M,Nc) = [relu](A(M,K) @ Bt(Nc,K)^T [+ bias]); A,Bt bf16 row-major.
// Block: 128 threads (4 wave32), tile 32(M) x 128(N), BK=32, double-buffered.
// Wave w: M-subtile (w&1)*16, N-strip (w>>1)*64 -> 4 x 16x16 tiles (4 WMMA).
template <bool RELU_BIAS>
__global__ __launch_bounds__(128) void gemm_bf16_wmma(
    const bf16_t* __restrict__ A, const bf16_t* __restrict__ Bt,
    const float* __restrict__ bias, float* __restrict__ C,
    int M, int K, int Nc)
{
    __shared__ __align__(16) bf16_t lds_a[2][32 * 32];    // [m][k]
    __shared__ __align__(16) bf16_t lds_b[2][128 * 32];   // [n][k]

    const int tid    = threadIdx.x;
    const int lane   = tid & 31;
    const int waveId = tid >> 5;
    const int half   = lane >> 4;
    const int l15    = lane & 15;
    const int mBase  = blockIdx.y * 32;
    const int nBase  = blockIdx.x * 128;
    const int mSub   = waveId & 1;    // which 16-row subtile
    const int nSub   = waveId >> 1;   // which 64-col strip

    // Per-thread staging assignments (pure 16B copies, 5 async issues/stage):
    const int aRow  = tid >> 2;          // 0..31
    const int aKoff = (tid & 3) * 8;     // 0,8,16,24

    // Stage tile for K offset k0 into buffer buf (1 A chunk + 4 B chunks).
    auto stage = [&](int buf, int k0) {
        copy16(&A[(size_t)(mBase + aRow) * K + k0 + aKoff],
               &lds_a[buf][aRow * 32 + aKoff]);
#pragma unroll
        for (int it = 0; it < 4; ++it) {
            int c    = tid + it * 128;
            int row  = c >> 2;
            int koff = (c & 3) * 8;
            copy16(&Bt[(size_t)(nBase + row) * K + k0 + koff],
                   &lds_b[buf][row * 32 + koff]);
        }
    };

    v8f acc[4] = {};
    const int nsteps = K >> 5;

    stage(0, 0);                               // prologue: fill buffer 0
    for (int s = 0; s < nsteps; ++s) {
        const int cur = s & 1;
        const bool hasNext = (s + 1) < nsteps;
        if (hasNext) {
            stage(cur ^ 1, (s + 1) << 5);      // next tile streams during compute
            copy_wait<5>();                    // drain only current tile's 5 loads
        } else {
            copy_wait<0>();
        }
        __syncthreads();

        const bf16_t* la = lds_a[cur];
        const bf16_t* lb = lds_b[cur];
        // ---- fragments: 1 A frag reused across 4 B frags -> 4 WMMAs ----
        Frag fa;
        int mRow = mSub * 16 + l15;
        // A 16x32 bf16 layout: lane half interleaves K octets per VGPR group
        fa.f[0] = *reinterpret_cast<const float4*>(&la[mRow * 32 + half * 8]);
        fa.f[1] = *reinterpret_cast<const float4*>(&la[mRow * 32 + 16 + half * 8]);
#pragma unroll
        for (int t = 0; t < 4; ++t) {
            // B layout: lane half interleaves K octets per VGPR group (same as A)
            Frag fb;
            int nRow = nSub * 64 + t * 16 + l15;
            fb.f[0] = *reinterpret_cast<const float4*>(&lb[nRow * 32 + half * 8]);
            fb.f[1] = *reinterpret_cast<const float4*>(&lb[nRow * 32 + 16 + half * 8]);
            acc[t] = __builtin_amdgcn_wmma_f32_16x16x32_bf16(
                false, fa.v, false, fb.v, (short)0, acc[t], false, false);
        }
        __syncthreads();   // protect buffer cur before it is restaged at s+2
    }

    // ---- epilogue: lane half -> M+8, VGPR r -> M row, lane&15 -> N ----
#pragma unroll
    for (int t = 0; t < 4; ++t) {
        int col = nBase + nSub * 64 + t * 16 + l15;
        float bcol = RELU_BIAS ? bias[col] : 0.f;
#pragma unroll
        for (int r = 0; r < 8; ++r) {
            int m = mBase + mSub * 16 + half * 8 + r;
            float vv = acc[t][r];
            if (RELU_BIAS) {
                vv += bcol;
                vv = vv > 0.f ? vv : 0.f;
            }
            C[(size_t)m * Nc + col] = vv;
        }
    }
}

// Packed fp32 -> bf16 convert (same layout). 2 elements/thread.
__global__ __launch_bounds__(256) void convert_kernel(
    const float* __restrict__ src, bf16_t* __restrict__ dst)
{
    int idx = (blockIdx.x * 256 + threadIdx.x) * 2;
    bf16x2 p = { (bf16_t)src[idx], (bf16_t)src[idx + 1] };
    *reinterpret_cast<bf16x2*>(&dst[idx]) = p;
}

// W (K x Nc fp32, row-major) -> Wt (Nc x K bf16, row-major), 32x32 LDS tiles.
__global__ __launch_bounds__(256) void transpose_convert_kernel(
    const float* __restrict__ W, bf16_t* __restrict__ Wt, int K, int Nc)
{
    __shared__ float tile[32][33];
    const int k0 = blockIdx.y * 32;
    const int n0 = blockIdx.x * 32;
    const int t  = threadIdx.x;
    const int c  = t & 31;            // n offset (coalesced reads)
    const int r0 = t >> 5;            // 0..7
#pragma unroll
    for (int i = 0; i < 4; ++i) {
        int r = r0 + i * 8;
        tile[r][c] = W[(size_t)(k0 + r) * Nc + n0 + c];
    }
    __syncthreads();
    const int n   = t >> 3;           // 0..31
    const int kp0 = t & 7;            // k-pair id
#pragma unroll
    for (int i = 0; i < 2; ++i) {
        int kp = kp0 + i * 8;         // 0..15
        bf16x2 p = { (bf16_t)tile[2 * kp][n], (bf16_t)tile[2 * kp + 1][n] };
        *reinterpret_cast<bf16x2*>(&Wt[(size_t)(n0 + n) * K + k0 + 2 * kp]) = p;
    }
}

// u[n,i,d] = bf16(v[n,i,d] * qe[n,d]); packed pair stores.
__global__ __launch_bounds__(256) void scale_kernel(
    const float* __restrict__ v, const float* __restrict__ qe,
    bf16_t* __restrict__ u)
{
    int idx = (blockIdx.x * 256 + threadIdx.x) * 2;    // exact coverage
    int d = idx & (DD - 1);
    int n = idx / (BB * DD);
    bf16x2 p = { (bf16_t)(v[idx] * qe[n * DD + d]),
                 (bf16_t)(v[idx + 1] * qe[n * DD + d + 1]) };
    *reinterpret_cast<bf16x2*>(&u[idx]) = p;
}

// x[n,i,d] = bf16( sum_j relu(A[n,i,d] + A[n,j,d] + b1[d]) )
__global__ __launch_bounds__(256) void pairwise_kernel(
    const float* __restrict__ Alin, const float* __restrict__ b1,
    bf16_t* __restrict__ x)
{
    int d = blockIdx.x * 256 + threadIdx.x;
    int n = blockIdx.y;
    const float bb = b1[d];
    const float* base = Alin + (size_t)n * BB * DD + d;
    float a[BB];
#pragma unroll
    for (int j = 0; j < BB; ++j) a[j] = base[j * DD];
    bf16_t* xb = x + (size_t)n * BB * DD + d;
    for (int i = 0; i < BB; ++i) {
        float ai = base[i * DD];     // re-load (cached) so a[] stays register-indexed
        float s = 0.f;
#pragma unroll
        for (int j = 0; j < BB; ++j) {
            float t = ai + a[j] + bb;
            s += t > 0.f ? t : 0.f;
        }
        xb[i * DD] = (bf16_t)s;
    }
}

extern "C" void kernel_launch(void* const* d_in, const int* in_sizes, int n_in,
                              void* d_out, int out_size, void* d_ws, size_t ws_size,
                              hipStream_t stream) {
    const float* v  = (const float*)d_in[0];
    const float* q  = (const float*)d_in[1];
    const float* W1 = (const float*)d_in[2];
    const float* b1 = (const float*)d_in[3];
    const float* W2 = (const float*)d_in[4];
    const float* b2 = (const float*)d_in[5];
    const float* W3 = (const float*)d_in[6];
    const float* b3 = (const float*)d_in[7];
    float* out = (float*)d_out;

    // Workspace layout (all 16B-aligned by construction)
    float*  qe   = (float*)d_ws;                        // 32*2048 fp32
    bf16_t* u    = (bf16_t*)(qe + (size_t)NN * DD);     // 1152*2048 bf16
    bf16_t* x    = u;                                   // reuse (u dead after Alin)
    float*  Alin = (float*)(u + (size_t)NN * BB * DD);  // 1152*2048 fp32
    bf16_t* qb   = (bf16_t*)(Alin + (size_t)NN * BB * DD); // 32*1024 bf16
    bf16_t* W1t  = qb + (size_t)NN * QQ;                // 2048*2048 bf16
    bf16_t* W2t  = W1t + (size_t)DD * DD;
    bf16_t* W3t  = W2t + (size_t)DD * DD;               // 2048*1024 bf16

    // 0) one-time bf16 conversions / transposes
    convert_kernel<<<(NN * QQ) / 512, 256, 0, stream>>>(q, qb);
    transpose_convert_kernel<<<dim3(DD / 32, QQ / 32), 256, 0, stream>>>(W3, W3t, QQ, DD);
    transpose_convert_kernel<<<dim3(DD / 32, DD / 32), 256, 0, stream>>>(W1, W1t, DD, DD);
    transpose_convert_kernel<<<dim3(DD / 32, DD / 32), 256, 0, stream>>>(W2, W2t, DD, DD);

    // 1) qe = relu(q @ W3 + b3)   (M=32, K=1024, N=2048)
    gemm_bf16_wmma<true><<<dim3(DD / 128, NN / 32), 128, 0, stream>>>(
        qb, W3t, b3, qe, NN, QQ, DD);

    // 2) u = bf16(v * qe)
    scale_kernel<<<(NN * BB * DD) / 512, 256, 0, stream>>>(v, qe, u);

    // 3) Alin = u @ W1            (M=1152, K=2048, N=2048), no bias/relu
    gemm_bf16_wmma<false><<<dim3(DD / 128, (NN * BB) / 32), 128, 0, stream>>>(
        u, W1t, nullptr, Alin, NN * BB, DD, DD);

    // 4) x = bf16(pairwise relu-reduce)
    pairwise_kernel<<<dim3(DD / 256, NN), 256, 0, stream>>>(Alin, b1, x);

    // 5) out = relu(x @ W2 + b2)  (M=1152, K=2048, N=2048)
    gemm_bf16_wmma<true><<<dim3(DD / 128, (NN * BB) / 32), 128, 0, stream>>>(
        x, W2t, b2, out, NN * BB, DD, DD);
}
